// LSTMSeq2SeqAttnModel_22213570855408
// MI455X (gfx1250) — compile-verified, matrix operand order checked by reading, not run on hardware
//
#include <hip/hip_runtime.h>

// ---------------------------------------------------------------------------
// LSTM seq2seq + attention for MI455X (gfx1250, wave32, WMMA f16 16x16x32)
// B=2048 T=256 D=10 H=128 4H=512 EMB=16 OUT=15
// ---------------------------------------------------------------------------

#define B_   2048
#define T_   256
#define H_   128
#define G4_  512
#define DIN_ 10
#define EMB_ 16
#define OUT_ 15
#define KD_  288   // padded decoder K: [h(128) | ctx(128) | emb(16) | pad(16)]

typedef _Float16 half_t;
typedef __attribute__((ext_vector_type(16))) _Float16 v16h;
typedef __attribute__((ext_vector_type(8)))  _Float16 v8h;
typedef __attribute__((ext_vector_type(8)))  float    v8f;

#define DEVI __device__ __forceinline__

DEVI v8f wmma_f16(v16h a, v16h b, v8f c) {
  return __builtin_amdgcn_wmma_f32_16x16x32_f16(false, a, false, b,
                                                (short)0, c, false, false);
}

// Async DMA one 16B chunk global -> LDS (tracked by ASYNCcnt).
// VDST operand carries the LDS byte address (hardware uses addr[31:0] of the
// generic shared pointer), VADDR the 64-bit global address.
DEVI void async_g2l_b128(unsigned int lds_addr, const void* g) {
  asm volatile("global_load_async_to_lds_b128 %0, %1, off"
               :: "v"(lds_addr), "v"((unsigned long long)(uintptr_t)g)
               : "memory");
}
DEVI void wait_async0() {
  asm volatile("s_wait_asynccnt 0x0" ::: "memory");
}

// A/B fragment for v_wmma_f32_16x16x32_f16 from a row-major row (K halves).
// ISA layout: lane's 16 halves = run of 8 at kbase, run of 8 at kbase+16,
// kbase = kt*32 + (lane>=16 ? 8 : 0).  Both runs are 16B-aligned b128 loads.
DEVI v16h fragK(const half_t* rowbase, int kt, int lane) {
  const half_t* p = rowbase + kt * 32 + ((lane >> 4) & 1) * 8;
  v8h lo = *(const v8h*)p;
  v8h hi = *(const v8h*)(p + 16);
  v16h r;
#pragma unroll
  for (int i = 0; i < 8; ++i) { r[i] = lo[i]; r[i + 8] = hi[i]; }
  return r;
}

// Same fragment but converting from an f32 row on the fly.
DEVI v16h fragK_f32(const float* rowbase, int kt, int lane) {
  const float* p = rowbase + kt * 32 + ((lane >> 4) & 1) * 8;
  v16h r;
#pragma unroll
  for (int i = 0; i < 8; ++i) {
    r[i]     = (half_t)p[i];
    r[i + 8] = (half_t)p[16 + i];
  }
  return r;
}

DEVI float sigm(float x) { return 1.f / (1.f + __expf(-x)); }

// ---------------------------------------------------------------------------
// Kernel 0: pack/convert weights to f16, B-matrix layout = W row-major [N][K]
// ---------------------------------------------------------------------------
__global__ __launch_bounds__(256) void k_prep(
    const float* __restrict__ Whh0, const float* __restrict__ Wih1,
    const float* __restrict__ Whh1, const float* __restrict__ Wq,
    const float* __restrict__ Wk,   const float* __restrict__ Wih_d,
    const float* __restrict__ Whh_d,
    half_t* __restrict__ W0, half_t* __restrict__ W1,
    half_t* __restrict__ Wq16, half_t* __restrict__ Wk16,
    half_t* __restrict__ Wd) {
  int i = blockIdx.x * blockDim.x + threadIdx.x;
  if (i < G4_ * H_) W0[i] = (half_t)Whh0[i];
  if (i < G4_ * 2 * H_) {
    int n = i >> 8, k = i & 255;
    W1[i] = (half_t)(k < H_ ? Wih1[n * H_ + k] : Whh1[n * H_ + (k - H_)]);
  }
  if (i < H_ * H_) { Wq16[i] = (half_t)Wq[i]; Wk16[i] = (half_t)Wk[i]; }
  if (i < G4_ * KD_) {
    int n = i / KD_, k = i % KD_;
    float v = 0.f;
    if (k < 128)      v = Whh_d[n * H_ + k];            // h part
    else if (k < 272) v = Wih_d[n * 144 + (k - 128)];   // ctx(128)+emb(16)
    Wd[i] = (half_t)v;
  }
}

// ---------------------------------------------------------------------------
// Kernel 1: encoder layer 0. Persistent over t. 32 blocks x 128 thr (4 waves,
// 64 batch rows/block, 16 rows/wave). Whh0 f16 B-matrix pinned in LDS (128KB,
// filled with async global->LDS DMA). K=10 input projection in VALU folded
// into the WMMA accumulator init.
// ---------------------------------------------------------------------------
__global__ __launch_bounds__(128) void k_enc0(
    const float* __restrict__ x, const float* __restrict__ Wih0,
    const float* __restrict__ bih0, const float* __restrict__ bhh0,
    const half_t* __restrict__ W0hh16, half_t* __restrict__ h0s) {
  extern __shared__ char smem[];
  half_t* Wl = (half_t*)smem;                              // [512][128] 128KB
  half_t* hs = (half_t*)(smem + G4_ * H_ * 2);             // 4x[16][128]
  float*  xb = (float*)(smem + G4_ * H_ * 2 + 4 * 16 * H_ * 2); // [64][10]

  const int tid = threadIdx.x, lane = tid & 31, w = tid >> 5;
  const int b0 = blockIdx.x * 64;
  const int ln = lane & 15, g = lane >> 4;

  { // async DMA weight load into LDS
    unsigned int ldsb = (unsigned int)(uintptr_t)Wl;
    for (int i = tid; i < (G4_ * H_ * 2) / 16; i += blockDim.x)
      async_g2l_b128(ldsb + i * 16, (const char*)W0hh16 + (size_t)i * 16);
    wait_async0();
  }
  half_t* myhs = hs + w * 16 * H_;
  for (int i = lane; i < 16 * H_; i += 32) myhs[i] = (half_t)0.f;

  float biasv[32];
#pragma unroll
  for (int nt = 0; nt < 32; ++nt) {
    int n = nt * 16 + ln;
    biasv[nt] = bih0[n] + bhh0[n];
  }
  float c[8][8];
#pragma unroll
  for (int a = 0; a < 8; ++a)
#pragma unroll
    for (int r = 0; r < 8; ++r) c[a][r] = 0.f;
  __syncthreads();

  for (int t = 0; t < T_; ++t) {
    // stage x[b0..b0+63, t, 0:10] into LDS
    for (int i = tid; i < 64 * DIN_; i += blockDim.x) {
      int rr = i / DIN_, d = i % DIN_;
      xb[i] = x[(size_t)(b0 + rr) * T_ * DIN_ + (size_t)t * DIN_ + d];
    }
    __syncthreads();

    // A fragments (h_prev) - read before this step's staging writes
    v16h af[4];
#pragma unroll
    for (int kt = 0; kt < 4; ++kt) af[kt] = fragK(myhs + ln * H_, kt, lane);

#pragma unroll 1
    for (int jt = 0; jt < 8; ++jt) {      // 8 column tiles per gate
      v8f acc[4];
#pragma unroll
      for (int gi = 0; gi < 4; ++gi) {    // gates i,f,g,o
        int nt = jt + 8 * gi, n = nt * 16 + ln;
        float wv[DIN_];
#pragma unroll
        for (int d = 0; d < DIN_; ++d) wv[d] = Wih0[n * DIN_ + d];
#pragma unroll
        for (int r = 0; r < 8; ++r) {
          float s = biasv[nt];
          const float* xr = &xb[(w * 16 + g * 8 + r) * DIN_];
#pragma unroll
          for (int d = 0; d < DIN_; ++d) s += xr[d] * wv[d];
          acc[gi][r] = s;
        }
        const half_t* brow = Wl + n * H_;
#pragma unroll
        for (int kt = 0; kt < 4; ++kt)
          acc[gi] = wmma_f16(af[kt], fragK(brow, kt, lane), acc[gi]);
      }
      // LSTM cell update (all four gate values for a cell live in one lane)
#pragma unroll
      for (int r = 0; r < 8; ++r) {
        float iv = sigm(acc[0][r]), fv = sigm(acc[1][r]);
        float gv = tanhf(acc[2][r]), ov = sigm(acc[3][r]);
        float cn = fv * c[jt][r] + iv * gv;
        c[jt][r] = cn;
        float hv = ov * tanhf(cn);
        int m = g * 8 + r, j = jt * 16 + ln;
        myhs[m * H_ + j] = (half_t)hv;
        h0s[((size_t)(b0 + w * 16 + m) * T_ + t) * H_ + j] = (half_t)hv;
      }
    }
    __syncthreads();
  }
}

// ---------------------------------------------------------------------------
// Kernel 2: encoder layer 1. Fuses input GEMM + recurrent GEMM as one K=256
// WMMA against [Wih1;Whh1] pinned in LDS (256KB, async DMA). Writes enc_out
// f32 and final (h_n, c_n).
// ---------------------------------------------------------------------------
__global__ __launch_bounds__(128) void k_enc1(
    const half_t* __restrict__ h0s, const float* __restrict__ bih1,
    const float* __restrict__ bhh1, const half_t* __restrict__ W1cat,
    float* __restrict__ enc, float* __restrict__ hN, float* __restrict__ cN) {
  extern __shared__ char smem[];
  half_t* Wl = (half_t*)smem;                         // [512][256] 256KB
  half_t* hs = (half_t*)(smem + G4_ * 256 * 2);       // 4x[16][128]
  const int tid = threadIdx.x, lane = tid & 31, w = tid >> 5;
  const int b0 = blockIdx.x * 64;
  const int ln = lane & 15, g = lane >> 4;

  {
    unsigned int ldsb = (unsigned int)(uintptr_t)Wl;
    for (int i = tid; i < (G4_ * 256 * 2) / 16; i += blockDim.x)
      async_g2l_b128(ldsb + i * 16, (const char*)W1cat + (size_t)i * 16);
    wait_async0();
  }
  half_t* myhs = hs + w * 16 * H_;
  for (int i = lane; i < 16 * H_; i += 32) myhs[i] = (half_t)0.f;

  float biasv[32];
#pragma unroll
  for (int nt = 0; nt < 32; ++nt) {
    int n = nt * 16 + ln;
    biasv[nt] = bih1[n] + bhh1[n];
  }
  float c[8][8];
#pragma unroll
  for (int a = 0; a < 8; ++a)
#pragma unroll
    for (int r = 0; r < 8; ++r) c[a][r] = 0.f;
  __syncthreads();

  for (int t = 0; t < T_; ++t) {
    v16h af[8];
    const half_t* arow = &h0s[((size_t)(b0 + w * 16 + ln) * T_ + t) * H_];
#pragma unroll
    for (int kt = 0; kt < 4; ++kt) af[kt] = fragK(arow, kt, lane);       // x_t
#pragma unroll
    for (int kt = 0; kt < 4; ++kt) af[4 + kt] = fragK(myhs + ln * H_, kt, lane); // h_{t-1}

#pragma unroll 1
    for (int jt = 0; jt < 8; ++jt) {
      v8f acc[4];
#pragma unroll
      for (int gi = 0; gi < 4; ++gi) {
        int nt = jt + 8 * gi, n = nt * 16 + ln;
#pragma unroll
        for (int r = 0; r < 8; ++r) acc[gi][r] = biasv[nt];
        const half_t* brow = Wl + n * 256;
#pragma unroll
        for (int kt = 0; kt < 8; ++kt)
          acc[gi] = wmma_f16(af[kt], fragK(brow, kt, lane), acc[gi]);
      }
#pragma unroll
      for (int r = 0; r < 8; ++r) {
        float iv = sigm(acc[0][r]), fv = sigm(acc[1][r]);
        float gv = tanhf(acc[2][r]), ov = sigm(acc[3][r]);
        float cn = fv * c[jt][r] + iv * gv;
        c[jt][r] = cn;
        float hv = ov * tanhf(cn);
        int m = g * 8 + r, j = jt * 16 + ln;
        size_t row = (size_t)(b0 + w * 16 + m);
        myhs[m * H_ + j] = (half_t)hv;
        enc[(row * T_ + t) * H_ + j] = hv;
        if (t == T_ - 1) {
          hN[row * H_ + j] = hv;
          cN[row * H_ + j] = cn;
        }
      }
    }
  }
}

// ---------------------------------------------------------------------------
// Kernel 3: keys = enc_out @ Wk^T  (M = B*T = 524288, K = N = 128), f16 out.
// 4096 blocks x 256 thr, one 16-row M tile per wave, Wk f16 in LDS.
// ---------------------------------------------------------------------------
__global__ __launch_bounds__(256) void k_keys(
    const float* __restrict__ enc, const half_t* __restrict__ Wk16,
    half_t* __restrict__ keys) {
  __shared__ half_t Wl[H_ * H_];
  const int tid = threadIdx.x, lane = tid & 31, w = tid >> 5;
  const int ln = lane & 15, g = lane >> 4;
  {
    unsigned int ldsb = (unsigned int)(uintptr_t)&Wl[0];
    for (int i = tid; i < (H_ * H_ * 2) / 16; i += blockDim.x)
      async_g2l_b128(ldsb + i * 16, (const char*)Wk16 + (size_t)i * 16);
    wait_async0();
  }
  __syncthreads();

  size_t R = ((size_t)blockIdx.x * 8 + w) * 16;
  v16h af[4];
  const float* arow = &enc[(R + ln) * H_];
#pragma unroll
  for (int kt = 0; kt < 4; ++kt) af[kt] = fragK_f32(arow, kt, lane);

#pragma unroll 1
  for (int nt = 0; nt < 8; ++nt) {
    v8f acc;
#pragma unroll
    for (int r = 0; r < 8; ++r) acc[r] = 0.f;
    const half_t* brow = Wl + (nt * 16 + ln) * H_;
#pragma unroll
    for (int kt = 0; kt < 4; ++kt)
      acc = wmma_f16(af[kt], fragK(brow, kt, lane), acc);
#pragma unroll
    for (int r = 0; r < 8; ++r)
      keys[(R + g * 8 + r) * H_ + nt * 16 + ln] = (half_t)acc[r];
  }
}

// ---------------------------------------------------------------------------
// Kernel 4: decoder, persistent over the 15 steps. 32 blocks x 128 thr.
// Attention is the bandwidth floor (keys f16 + enc f32 re-streamed 15x):
// all streaming loads are b128-vectorized with global_prefetch ahead.
// ---------------------------------------------------------------------------
__global__ __launch_bounds__(128) void k_dec(
    const half_t* __restrict__ keys, const float* __restrict__ enc,
    const float* __restrict__ hN, const float* __restrict__ cN,
    const half_t* __restrict__ Wq16, const half_t* __restrict__ Wd16,
    const float* __restrict__ bih_d, const float* __restrict__ bhh_d,
    const float* __restrict__ v_attn, const float* __restrict__ step_emb,
    const float* __restrict__ ln_g, const float* __restrict__ ln_b,
    const float* __restrict__ Wo, const float* __restrict__ bo,
    float* __restrict__ out) {
  extern __shared__ char smem[];
  // per-wave scratch: gateA f16 [16][288] | qbuf f32 [16][128] | sbuf f32 [16][256]
  const int PW = 16 * KD_ * 2 + 16 * H_ * 4 + 16 * T_ * 4;   // 33792 B
  const int tid = threadIdx.x, lane = tid & 31, w = tid >> 5;
  const int b0 = blockIdx.x * 64;
  const int ln = lane & 15, g = lane >> 4;
  char* base = smem + w * PW;
  half_t* gateA = (half_t*)base;
  float* qbuf = (float*)(base + 16 * KD_ * 2);
  float* sbuf = (float*)(base + 16 * KD_ * 2 + 16 * H_ * 4);
  float* sv   = (float*)(smem + 4 * PW);  // [v(128) | ln_g(128) | ln_b(128) | Wo(128)]

  for (int i = tid; i < H_; i += blockDim.x) {
    sv[i] = v_attn[i];
    sv[H_ + i] = ln_g[i];
    sv[2 * H_ + i] = ln_b[i];
    sv[3 * H_ + i] = Wo[i];
  }
  float biasv[32];
#pragma unroll
  for (int nt = 0; nt < 32; ++nt) {
    int n = nt * 16 + ln;
    biasv[nt] = bih_d[n] + bhh_d[n];
  }
  float h[8][8], c[8][8];
#pragma unroll
  for (int jt = 0; jt < 8; ++jt)
#pragma unroll
    for (int r = 0; r < 8; ++r) {
      size_t row = (size_t)(b0 + w * 16 + g * 8 + r);
      h[jt][r] = hN[row * H_ + jt * 16 + ln];
      c[jt][r] = cN[row * H_ + jt * 16 + ln];
    }
  const float bov = bo[0];
  __syncthreads();

  for (int s = 0; s < OUT_; ++s) {
    // ---- stage h into gateA[:,0:128], emb into [:,256:288] ----
#pragma unroll
    for (int jt = 0; jt < 8; ++jt)
#pragma unroll
      for (int r = 0; r < 8; ++r)
        gateA[(g * 8 + r) * KD_ + jt * 16 + ln] = (half_t)h[jt][r];
    if (lane < 16) {
#pragma unroll
      for (int e = 0; e < EMB_; ++e)
        gateA[lane * KD_ + 256 + e] = (half_t)step_emb[s * EMB_ + e];
#pragma unroll
      for (int e = EMB_; e < 32; ++e)
        gateA[lane * KD_ + 256 + e] = (half_t)0.f;
    }
    __syncthreads();

    // ---- q = h @ Wq^T (WMMA) -> qbuf ----
#pragma unroll 1
    for (int nt = 0; nt < 8; ++nt) {
      v8f acc;
#pragma unroll
      for (int r = 0; r < 8; ++r) acc[r] = 0.f;
      const half_t* arow = gateA + ln * KD_;
      const half_t* brow = Wq16 + (nt * 16 + ln) * H_;
#pragma unroll
      for (int kt = 0; kt < 4; ++kt)
        acc = wmma_f16(fragK(arow, kt, lane), fragK(brow, kt, lane), acc);
#pragma unroll
      for (int r = 0; r < 8; ++r)
        qbuf[(g * 8 + r) * H_ + nt * 16 + ln] = acc[r];
    }
    __syncthreads();

    // ---- attention: lane handles (row = ln, t-half = g) ----
    const int mrow = ln, th = g;
    const size_t grow = (size_t)(b0 + w * 16 + mrow);
    const float4* qrow = (const float4*)(qbuf + mrow * H_);
    const float4* vrow = (const float4*)sv;
    float mx = -1e30f;
    for (int tt = 0; tt < 128; ++tt) {
      int t = th * 128 + tt;
      const v8h* kr = (const v8h*)&keys[(grow * T_ + t) * H_];
      if (tt + 8 < 128)
        __builtin_prefetch(&keys[(grow * T_ + t + 8) * H_], 0, 1);
      float sc = 0.f;
#pragma unroll 4
      for (int jv = 0; jv < 16; ++jv) {
        v8h kv = kr[jv];
        float4 qa = qrow[2 * jv], qb = qrow[2 * jv + 1];
        float4 va = vrow[2 * jv], vb = vrow[2 * jv + 1];
        sc += va.x * tanhf(qa.x + (float)kv[0]);
        sc += va.y * tanhf(qa.y + (float)kv[1]);
        sc += va.z * tanhf(qa.z + (float)kv[2]);
        sc += va.w * tanhf(qa.w + (float)kv[3]);
        sc += vb.x * tanhf(qb.x + (float)kv[4]);
        sc += vb.y * tanhf(qb.y + (float)kv[5]);
        sc += vb.z * tanhf(qb.z + (float)kv[6]);
        sc += vb.w * tanhf(qb.w + (float)kv[7]);
      }
      sbuf[mrow * T_ + t] = sc;
      mx = fmaxf(mx, sc);
    }
    mx = fmaxf(mx, __shfl_xor(mx, 16, 32));
    float se = 0.f;
    for (int tt = 0; tt < 128; ++tt) {
      int t = th * 128 + tt;
      float e = __expf(sbuf[mrow * T_ + t] - mx);
      se += e;
      sbuf[mrow * T_ + t] = e;             // unnormalized attn weight
    }
    float stot = se + __shfl_xor(se, 16, 32);
    float inv = 1.f / stot;

    // ---- ctx = softmax(score) @ enc, j-chunked (32 reg accumulators),
    //      t-inner with float4 streaming so enc is touched exactly once ----
#pragma unroll 1
    for (int jc = 0; jc < 4; ++jc) {
      float a[32];
#pragma unroll
      for (int i = 0; i < 32; ++i) a[i] = 0.f;
      for (int tt = 0; tt < 128; ++tt) {
        int t = th * 128 + tt;
        float e = sbuf[mrow * T_ + t];
        const float4* er = (const float4*)&enc[(grow * T_ + t) * H_ + jc * 32];
        if (tt + 4 < 128)
          __builtin_prefetch(&enc[(grow * T_ + t + 4) * H_ + jc * 32], 0, 1);
#pragma unroll
        for (int v = 0; v < 8; ++v) {
          float4 q4 = er[v];
          a[v * 4 + 0] += e * q4.x;
          a[v * 4 + 1] += e * q4.y;
          a[v * 4 + 2] += e * q4.z;
          a[v * 4 + 3] += e * q4.w;
        }
      }
#pragma unroll
      for (int i = 0; i < 32; ++i) {
        float tot = a[i] + __shfl_xor(a[i], 16, 32);
        if (th == 0)
          gateA[mrow * KD_ + 128 + jc * 32 + i] = (half_t)(tot * inv);
      }
    }
    __syncthreads();

    // ---- gates = [h|ctx|emb] @ [Whh_d|Wih_d]^T + bias (WMMA, K=288) ----
    v16h af[9];
    const half_t* arow = gateA + ln * KD_;
#pragma unroll
    for (int kt = 0; kt < 9; ++kt) af[kt] = fragK(arow, kt, lane);
#pragma unroll 1
    for (int jt = 0; jt < 8; ++jt) {
      v8f acc[4];
#pragma unroll
      for (int gi = 0; gi < 4; ++gi) {
        int nt = jt + 8 * gi, n = nt * 16 + ln;
#pragma unroll
        for (int r = 0; r < 8; ++r) acc[gi][r] = biasv[nt];
        const half_t* brow = Wd16 + (size_t)n * KD_;
#pragma unroll
        for (int kt = 0; kt < 9; ++kt)
          acc[gi] = wmma_f16(af[kt], fragK(brow, kt, lane), acc[gi]);
      }
#pragma unroll
      for (int r = 0; r < 8; ++r) {
        float iv = sigm(acc[0][r]), fv = sigm(acc[1][r]);
        float gv = tanhf(acc[2][r]), ov = sigm(acc[3][r]);
        float cn = fv * c[jt][r] + iv * gv;
        c[jt][r] = cn;
        h[jt][r] = ov * tanhf(cn);
      }
    }

    // ---- layernorm + linear head; 16-lane xor reductions per row ----
#pragma unroll 1
    for (int r = 0; r < 8; ++r) {
      float sm = 0.f;
#pragma unroll
      for (int jt = 0; jt < 8; ++jt) sm += h[jt][r];
      sm += __shfl_xor(sm, 1, 32); sm += __shfl_xor(sm, 2, 32);
      sm += __shfl_xor(sm, 4, 32); sm += __shfl_xor(sm, 8, 32);
      float mu = sm * (1.f / 128.f);
      float sq = 0.f;
#pragma unroll
      for (int jt = 0; jt < 8; ++jt) { float d = h[jt][r] - mu; sq += d * d; }
      sq += __shfl_xor(sq, 1, 32); sq += __shfl_xor(sq, 2, 32);
      sq += __shfl_xor(sq, 4, 32); sq += __shfl_xor(sq, 8, 32);
      float rstd = rsqrtf(sq * (1.f / 128.f) + 1e-5f);
      float pp = 0.f;
#pragma unroll
      for (int jt = 0; jt < 8; ++jt) {
        int j = jt * 16 + ln;
        float hn = (h[jt][r] - mu) * rstd * sv[H_ + j] + sv[2 * H_ + j];
        pp += hn * sv[3 * H_ + j];
      }
      pp += __shfl_xor(pp, 1, 32); pp += __shfl_xor(pp, 2, 32);
      pp += __shfl_xor(pp, 4, 32); pp += __shfl_xor(pp, 8, 32);
      if (ln == 0)
        out[(size_t)(b0 + w * 16 + g * 8 + r) * OUT_ + s] = pp + bov;
    }
    __syncthreads();
  }
}

// ---------------------------------------------------------------------------
// Host launcher
// ---------------------------------------------------------------------------
extern "C" void kernel_launch(void* const* d_in, const int* in_sizes, int n_in,
                              void* d_out, int out_size, void* d_ws,
                              size_t ws_size, hipStream_t stream) {
  const float* x      = (const float*)d_in[0];
  const float* Wih0   = (const float*)d_in[1];
  const float* Whh0   = (const float*)d_in[2];
  const float* bih0   = (const float*)d_in[3];
  const float* bhh0   = (const float*)d_in[4];
  const float* Wih1   = (const float*)d_in[5];
  const float* Whh1   = (const float*)d_in[6];
  const float* bih1   = (const float*)d_in[7];
  const float* bhh1   = (const float*)d_in[8];
  const float* Wq     = (const float*)d_in[9];
  const float* Wk     = (const float*)d_in[10];
  const float* v_attn = (const float*)d_in[11];
  const float* semb   = (const float*)d_in[12];
  const float* Wih_d  = (const float*)d_in[13];
  const float* Whh_d  = (const float*)d_in[14];
  const float* bih_d  = (const float*)d_in[15];
  const float* bhh_d  = (const float*)d_in[16];
  const float* ln_g   = (const float*)d_in[17];
  const float* ln_b   = (const float*)d_in[18];
  const float* Wo     = (const float*)d_in[19];
  const float* bo     = (const float*)d_in[20];
  float* out = (float*)d_out;

  char* ws = (char*)d_ws;
  size_t off = 0;
  auto take = [&](size_t bytes) {
    char* p = ws + off;
    off = (off + bytes + 255) & ~(size_t)255;
    return p;
  };
  half_t* W0    = (half_t*)take((size_t)G4_ * H_ * 2);
  half_t* W1    = (half_t*)take((size_t)G4_ * 256 * 2);
  half_t* Wq16  = (half_t*)take((size_t)H_ * H_ * 2);
  half_t* Wk16  = (half_t*)take((size_t)H_ * H_ * 2);
  half_t* Wd16  = (half_t*)take((size_t)G4_ * KD_ * 2);
  half_t* h0s   = (half_t*)take((size_t)B_ * T_ * H_ * 2);
  float*  encb  = (float*) take((size_t)B_ * T_ * H_ * 4);
  half_t* keysb = (half_t*)take((size_t)B_ * T_ * H_ * 2);
  float*  hNb   = (float*) take((size_t)B_ * H_ * 4);
  float*  cNb   = (float*) take((size_t)B_ * H_ * 4);

  // 0: pack weights to f16 [N][K]
  k_prep<<<(G4_ * KD_ + 255) / 256, 256, 0, stream>>>(
      Whh0, Wih1, Whh1, Wq, Wk, Wih_d, Whh_d, W0, W1, Wq16, Wk16, Wd16);

  // 1: encoder layer 0 (persistent, Whh0 in LDS: 128KB + staging + x tile)
  size_t lds0 = (size_t)G4_ * H_ * 2 + 4 * 16 * H_ * 2 + 64 * DIN_ * 4;
  k_enc0<<<B_ / 64, 128, lds0, stream>>>(x, Wih0, bih0, bhh0, W0, h0s);

  // 2: encoder layer 1 (persistent, [Wih1;Whh1] in LDS: 256KB + staging)
  size_t lds1 = (size_t)G4_ * 256 * 2 + 4 * 16 * H_ * 2;
  k_enc1<<<B_ / 64, 128, lds1, stream>>>(h0s, bih1, bhh1, W1, encb, hNb, cNb);

  // 3: keys = enc @ Wk^T (one 16-row tile per wave)
  k_keys<<<(B_ * T_) / (16 * 8), 256, 0, stream>>>(encb, Wk16, keysb);

  // 4: decoder (persistent over 15 steps)
  size_t ldsd = 4 * (16 * KD_ * 2 + 16 * H_ * 4 + 16 * T_ * 4) + 4 * H_ * 4;
  k_dec<<<B_ / 64, 128, ldsd, stream>>>(keysb, encb, hNb, cNb, Wq16, Wd16,
                                        bih_d, bhh_d, v_attn, semb, ln_g, ln_b,
                                        Wo, bo, out);
}
